// SelfAttention_75763223101596
// MI455X (gfx1250) — matrix operand order repaired and checked by reading, not compiled, not run on hardware
//
#include <hip/hip_runtime.h>

// MI455X / gfx1250 — wave32, WMMA 16x16x32 f16->f32, double-buffered async
// global->LDS DMA (ASYNCcnt), DPP16 butterfly softmax reductions.
// S=8192, E=1024, D=64. HBM-bound on 100MB fp32 activation reads; q/k/v kept
// fp16 in workspace (V and W pre-transposed so all staging is contiguous b128).
// 1/sqrt(S) folded into the Q projection; causal mask applied only on the
// (at most two) diagonal KV tiles per wave.

typedef __attribute__((ext_vector_type(16))) _Float16 v16h;
typedef __attribute__((ext_vector_type(8)))  float    v8f;

#define S_LEN 8192
#define E_DIM 1024
#define D_DIM 64

__device__ __forceinline__ v8f wmma16x16x32(v16h a, v16h b, v8f c) {
  return __builtin_amdgcn_wmma_f32_16x16x32_f16(false, a, false, b, (short)0, c,
                                                false, false);
}

// A-fragment element j -> K index (contiguous: k = 16*hi + j, matches B loads)
__device__ __forceinline__ int a_frag_k(int j, int hi) {
  return hi * 16 + j;
}

// ---- async DMA helpers (ASYNCcnt-tracked, in-order completion) -------------
__device__ __forceinline__ void async_b128(const _Float16* g, _Float16* l) {
  unsigned int       laddr = (unsigned int)(unsigned long long)l;  // LDS offset
  unsigned long long gaddr = (unsigned long long)g;
  asm volatile("global_load_async_to_lds_b128 %0, %1, off"
               :: "v"(laddr), "v"(gaddr) : "memory");
}
#define WAIT_ASYNC(n) asm volatile("s_wait_asynccnt " #n ::: "memory")

// ---- DPP16 butterfly reduction over each 16-lane group (pure VALU) ---------
template <int CTRL>
__device__ __forceinline__ float dpp_f32(float x) {
  int r = __builtin_amdgcn_mov_dpp(__builtin_bit_cast(int, x), CTRL, 0xf, 0xf,
                                   true);
  return __builtin_bit_cast(float, r);
}
__device__ __forceinline__ float red_max16(float x) {
  x = fmaxf(x, dpp_f32<0xB1>(x));   // quad_perm [1,0,3,2]  (xor 1)
  x = fmaxf(x, dpp_f32<0x4E>(x));   // quad_perm [2,3,0,1]  (xor 2)
  x = fmaxf(x, dpp_f32<0x141>(x));  // row_half_mirror      (merges quads)
  x = fmaxf(x, dpp_f32<0x140>(x));  // row_mirror           (merges halves)
  return x;
}
__device__ __forceinline__ float red_sum16(float x) {
  x += dpp_f32<0xB1>(x);
  x += dpp_f32<0x4E>(x);
  x += dpp_f32<0x141>(x);
  x += dpp_f32<0x140>(x);
  return x;
}

// Online-softmax tile update; MASK only instantiated for diagonal tiles.
template <bool MASK>
__device__ __forceinline__ void softmax_update(
    const v8f& s0, const v8f& s1, int kv, int qbase, int lo, int hi,
    float* m_r, float* l_r, v8f* o, _Float16* prow) {
  const float LOG2E = 1.4426950408889634f;
#pragma unroll
  for (int r = 0; r < 8; ++r) {
    float x0 = s0[r], x1 = s1[r];
    if (MASK) {
      int row = qbase + r + 8 * hi;
      x0 = (kv + lo      <= row) ? x0 : -__builtin_inff();
      x1 = (kv + 16 + lo <= row) ? x1 : -__builtin_inff();
    }
    float tmax  = red_max16(fmaxf(x0, x1));
    float mnew  = fmaxf(m_r[r], tmax);
    float alpha = exp2f((m_r[r] - mnew) * LOG2E);
    float p0 = exp2f((x0 - mnew) * LOG2E);
    float p1 = exp2f((x1 - mnew) * LOG2E);

    l_r[r] = l_r[r] * alpha + red_sum16(p0 + p1);
    m_r[r] = mnew;
#pragma unroll
    for (int nc = 0; nc < 4; ++nc) o[nc][r] *= alpha;

    prow[(r + 8 * hi) * 32 + lo]      = (_Float16)p0;
    prow[(r + 8 * hi) * 32 + 16 + lo] = (_Float16)p1;
  }
}

// ---------------------------------------------------------------------------
// One-shot: Wt[d][e] (fp16, [64][1024]) = (half)W[e][d].
// ---------------------------------------------------------------------------
__global__ __launch_bounds__(256)
void transposeW(const float* __restrict__ W, _Float16* __restrict__ Wt) {
  int o0 = (blockIdx.x * 256 + threadIdx.x) * 4;   // grid 64 -> 65536 elems
  int d  = o0 >> 10;
  int e0 = o0 & 1023;
#pragma unroll
  for (int i = 0; i < 4; ++i)
    Wt[o0 + i] = (_Float16)W[(size_t)(e0 + i) * D_DIM + d];
}

// ---------------------------------------------------------------------------
// Projection: Y = (X[S,1024] @ W[1024,64] + b) * out_scale, output fp16.
// TR=false: Y[row][d] (Q,K).  TR=true: Y[d][row] (V^T).
// 256 threads = 8 waves, 16 rows/wave. W chunks double-buffered via async DMA.
// ---------------------------------------------------------------------------
template <bool TR>
__global__ __launch_bounds__(256)
void proj16(const float* __restrict__ X, const _Float16* __restrict__ Wt_g,
            const float* __restrict__ bias, _Float16* __restrict__ Y,
            float out_scale) {
  __shared__ __align__(32) _Float16 Wt[2][64 * 32];   // [buf][d][e_local]

  const int t    = threadIdx.x;
  const int lane = t & 31;
  const int wv   = t >> 5;
  const int lo   = lane & 15;
  const int hi   = lane >> 4;
  const int row0 = blockIdx.x * 128 + wv * 16;

  v8f acc[4] = {};

  // prime buffer 0
  async_b128(Wt_g + (size_t)(t >> 2) * E_DIM + (t & 3) * 8, &Wt[0][t * 8]);

  for (int e = 0; e < E_DIM; e += 32) {
    const int  cur  = (e >> 5) & 1;
    const bool next = (e + 32 < E_DIM);
    if (next)  // issue next chunk's DMA before waiting on current
      async_b128(Wt_g + (size_t)(t >> 2) * E_DIM + (e + 32) + (t & 3) * 8,
                 &Wt[cur ^ 1][t * 8]);
    if (next) { WAIT_ASYNC(0x1); } else { WAIT_ASYNC(0x0); }
    __syncthreads();

    // A fragment of X rows (fp32 -> fp16 on the fly)
    v16h a;
#pragma unroll
    for (int j = 0; j < 16; ++j)
      a[j] = (_Float16)X[(size_t)(row0 + lo) * E_DIM + e + a_frag_k(j, hi)];

    v16h b[4];
#pragma unroll
    for (int nc = 0; nc < 4; ++nc)
      b[nc] = *(const v16h*)&Wt[cur][(nc * 16 + lo) * 32 + hi * 16];
#pragma unroll
    for (int nc = 0; nc < 4; ++nc)
      acc[nc] = wmma16x16x32(a, b[nc], acc[nc]);

    __syncthreads();   // everyone done with Wt[cur] before it is re-staged
  }

#pragma unroll
  for (int nc = 0; nc < 4; ++nc) {
    float bv = bias[nc * 16 + lo];
#pragma unroll
    for (int r = 0; r < 8; ++r) {
      int row = row0 + r + 8 * hi;
      _Float16 v = (_Float16)((acc[nc][r] + bv) * out_scale);
      if (TR) Y[(size_t)(nc * 16 + lo) * S_LEN + row] = v;  // contiguous in r
      else    Y[(size_t)row * D_DIM + nc * 16 + lo]   = v;
    }
  }
}

// ---------------------------------------------------------------------------
// Causal flash attention. 256 threads = 8 waves; wave owns 16 Q rows.
// KV tiles of 32 double-buffered via async b128 DMA (K row-major, V from
// pre-transposed VT). Online softmax fp32 with DPP16 reductions; scale is
// pre-folded into Q, masking only on diagonal tiles.
// ---------------------------------------------------------------------------
__global__ __launch_bounds__(256)
void fattn(const _Float16* __restrict__ qh, const _Float16* __restrict__ kh,
           const _Float16* __restrict__ vt, float* __restrict__ out) {
  __shared__ __align__(32) _Float16 Klds[2][32 * 64];   // [buf][kv_row][d]
  __shared__ __align__(32) _Float16 Vlds[2][64 * 32];   // [buf][d][kv_row]
  __shared__ __align__(32) _Float16 Pw[8][16 * 32];     // per-wave P scratch

  const int t    = threadIdx.x;
  const int lane = t & 31;
  const int wv   = t >> 5;
  const int lo   = lane & 15;
  const int hi   = lane >> 4;
  const int qbase   = blockIdx.x * 128 + wv * 16;
  const int qlast_b = blockIdx.x * 128 + 127;

  // Q fragments (two K=32 chunks over D=64), loaded once
  v16h aq[2];
#pragma unroll
  for (int c = 0; c < 2; ++c)
#pragma unroll
    for (int j = 0; j < 16; ++j)
      aq[c][j] = qh[(size_t)(qbase + lo) * D_DIM + c * 32 + a_frag_k(j, hi)];

  v8f o[4] = {};
  float m_r[8], l_r[8];
#pragma unroll
  for (int r = 0; r < 8; ++r) { m_r[r] = -__builtin_inff(); l_r[r] = 0.0f; }

  // prime buffer 0 with tile kv=0 (2 async ops per thread per tile)
  async_b128(kh + (size_t)t * 8, &Klds[0][t * 8]);
  async_b128(vt + (size_t)(t >> 2) * S_LEN + (t & 3) * 8, &Vlds[0][t * 8]);

  for (int kv = 0; kv <= qlast_b; kv += 32) {
    const int  cur  = (kv >> 5) & 1;
    const bool next = (kv + 32 <= qlast_b);
    if (next) {  // overlap next tile's DMA with this tile's compute
      async_b128(kh + (size_t)(kv + 32) * D_DIM + t * 8, &Klds[cur ^ 1][t * 8]);
      async_b128(vt + (size_t)(t >> 2) * S_LEN + (kv + 32) + (t & 3) * 8,
                 &Vlds[cur ^ 1][t * 8]);
    }
    if (next) { WAIT_ASYNC(0x2); } else { WAIT_ASYNC(0x0); }
    __syncthreads();

    if (kv <= qbase + 15) {            // wave-uniform causal tile skip
      // S = Q @ K^T (scale pre-folded into Q)
      v16h kb[4];
#pragma unroll
      for (int h = 0; h < 2; ++h)
#pragma unroll
        for (int c = 0; c < 2; ++c)
          kb[h * 2 + c] = *(const v16h*)
              &Klds[cur][(size_t)(h * 16 + lo) * 64 + c * 32 + hi * 16];
      v8f s0 = {}, s1 = {};
      s0 = wmma16x16x32(aq[0], kb[0], s0);
      s0 = wmma16x16x32(aq[1], kb[1], s0);
      s1 = wmma16x16x32(aq[0], kb[2], s1);
      s1 = wmma16x16x32(aq[1], kb[3], s1);

      // online softmax; mask chain only on diagonal tiles (wave-uniform)
      if (kv + 31 <= qbase)
        softmax_update<false>(s0, s1, kv, qbase, lo, hi, m_r, l_r, o, &Pw[wv][0]);
      else
        softmax_update<true>(s0, s1, kv, qbase, lo, hi, m_r, l_r, o, &Pw[wv][0]);

      // O += P @ V (same-wave LDS bounce re-lays P out as an A-fragment)
      v16h ap;
#pragma unroll
      for (int j = 0; j < 16; ++j)
        ap[j] = Pw[wv][lo * 32 + a_frag_k(j, hi)];
      v16h vb[4];
#pragma unroll
      for (int nc = 0; nc < 4; ++nc)
        vb[nc] = *(const v16h*)
            &Vlds[cur][(size_t)(nc * 16 + lo) * 32 + hi * 16];
#pragma unroll
      for (int nc = 0; nc < 4; ++nc)
        o[nc] = wmma16x16x32(ap, vb[nc], o[nc]);
    }

    __syncthreads();   // all waves done with buffer `cur` before re-staging
  }

#pragma unroll
  for (int nc = 0; nc < 4; ++nc)
#pragma unroll
    for (int r = 0; r < 8; ++r) {
      int row = qbase + r + 8 * hi;
      out[(size_t)row * D_DIM + nc * 16 + lo] = o[nc][r] / l_r[r];
    }
}

extern "C" void kernel_launch(void* const* d_in, const int* in_sizes, int n_in,
                              void* d_out, int out_size, void* d_ws, size_t ws_size,
                              hipStream_t stream) {
  const float* x  = (const float*)d_in[0];
  const float* y  = (const float*)d_in[1];
  const float* z  = (const float*)d_in[2];
  const float* Wq = (const float*)d_in[3];
  const float* bq = (const float*)d_in[4];
  const float* Wk = (const float*)d_in[5];
  const float* bk = (const float*)d_in[6];
  const float* Wv = (const float*)d_in[7];
  const float* bv = (const float*)d_in[8];

  _Float16* qh  = (_Float16*)d_ws;                      // [S][64], pre-scaled
  _Float16* kh  = qh  + (size_t)S_LEN * D_DIM;          // [S][64]
  _Float16* vtg = kh  + (size_t)S_LEN * D_DIM;          // [64][S]
  _Float16* wtq = vtg + (size_t)S_LEN * D_DIM;          // [64][1024] x3
  _Float16* wtk = wtq + (size_t)D_DIM * E_DIM;
  _Float16* wtv = wtk + (size_t)D_DIM * E_DIM;

  const float sc = 0.011048543456039806f;   // 1/sqrt(8192), folded into Q

  dim3 blk(256);
  hipLaunchKernelGGL(transposeW, dim3(64), blk, 0, stream, Wq, wtq);
  hipLaunchKernelGGL(transposeW, dim3(64), blk, 0, stream, Wk, wtk);
  hipLaunchKernelGGL(transposeW, dim3(64), blk, 0, stream, Wv, wtv);

  dim3 grd(S_LEN / 128);
  hipLaunchKernelGGL((proj16<false>), grd, blk, 0, stream, x, wtq, bq, qh, sc);
  hipLaunchKernelGGL((proj16<false>), grd, blk, 0, stream, y, wtk, bk, kh, 1.f);
  hipLaunchKernelGGL((proj16<true>),  grd, blk, 0, stream, z, wtv, bv, vtg, 1.f);
  hipLaunchKernelGGL(fattn, grd, blk, 0, stream, qh, kh, vtg, (float*)d_out);
}
